// NF4MLP_14121852469688
// MI455X (gfx1250) — matrix operand order repaired and checked by reading, not compiled, not run on hardware
//
#include <hip/hip_runtime.h>
#include <stdint.h>

// ---------------------------------------------------------------------------
// NF4 SwiGLU MLP for MI455X (gfx1250): fused dequant + bf16 WMMA GEMMs.
// Round 4: A-tiles now use double-buffered GLOBAL_LOAD_ASYNC_TO_LDS_B128
// (ASYNCcnt path) issued one K-tile ahead; weights stay register-staged
// through the VALU nf4 decode.
// ---------------------------------------------------------------------------

typedef __attribute__((ext_vector_type(16))) __bf16 v16bf;
typedef __attribute__((ext_vector_type(8)))  __bf16 v8bf;
typedef __attribute__((ext_vector_type(2)))  __bf16 v2bf;
typedef __attribute__((ext_vector_type(8)))  float  v8f;

#define T_  2048
#define D_  4096
#define H_  11008
#define NBLK ((H_ * D_) / 64)   // 704512 nf4 blocks per weight matrix

#define BM  256
#define BN  128
#define BK  64
#define BKP 72                   // padded LDS row stride (144B, 16B aligned)
#define NTHREADS 512             // 16 waves: 4 along M x 4 along N

__device__ __constant__ float NF4C[16] = {
    -1.0f, -0.6961928009986877f, -0.5250730514526367f, -0.39491748809814453f,
    -0.28444138169288635f, -0.18477343022823334f, -0.09105003625154495f, 0.0f,
    0.07958029955625534f, 0.16093020141124725f, 0.24611230194568634f,
    0.33791524171829224f, 0.44070982933044434f, 0.5626170039176941f,
    0.7229568362236023f, 1.0f};

// ---------------------------------------------------------------------------
// Async global -> LDS copy primitives (CDNA5 ASYNCcnt path)
// ---------------------------------------------------------------------------
static __device__ inline unsigned lds_off32(const void* p) {
  // generic -> LDS address space: pointer value becomes the 32-bit LDS offset
  return (unsigned)(uintptr_t)(__attribute__((address_space(3))) const void*)p;
}

static __device__ inline void async_load_b128(const void* gsrc, void* ldst) {
  asm volatile("global_load_async_to_lds_b128 %0, %1, off"
               :
               : "v"(lds_off32(ldst)), "v"((unsigned long long)(uintptr_t)gsrc)
               : "memory");
}

static __device__ inline void wait_async0() {
  asm volatile("s_wait_asynccnt 0" ::: "memory");
}

// ---------------------------------------------------------------------------
// Stage 0a: undo double quantization: scaler[b] = qs[b] / qf[b/256] + mean
// ---------------------------------------------------------------------------
__global__ __launch_bounds__(256) void nf4_scalers_kernel(
    const int8_t* __restrict__ qs, const float* __restrict__ qf,
    const float* __restrict__ mean, float* __restrict__ out, int n) {
  int i = blockIdx.x * 256 + threadIdx.x;
  if (i < n) out[i] = (float)qs[i] / qf[i >> 8] + mean[0];
}

// Convert 4 fp32 -> 4 bf16 packed (v_cvt_pk_bf16_f32 x2), one 8B store.
static __device__ inline void cvt4(float4 f, __bf16* dst) {
  union { unsigned u; v2bf v; } a, b;
  a.v[0] = (__bf16)f.x; a.v[1] = (__bf16)f.y;
  b.v[0] = (__bf16)f.z; b.v[1] = (__bf16)f.w;
  uint2 o; o.x = a.u; o.y = b.u;
  *(uint2*)dst = o;
}

// ---------------------------------------------------------------------------
// Stage 0b: x fp32 -> bf16 once (saves re-converting each x tile 86x).
// ---------------------------------------------------------------------------
__global__ __launch_bounds__(256) void x_to_bf16_kernel(
    const float* __restrict__ x, __bf16* __restrict__ xb, int n4) {
  int i = blockIdx.x * 256 + threadIdx.x;
  if (i < n4) cvt4(((const float4*)x)[i], xb + (size_t)i * 4);
}

// Build a v16bf fragment from two aligned 16-byte LDS chunks.
static __device__ inline v16bf frag2(const __bf16* p0, const __bf16* p1) {
  union { v16bf v; v8bf h[2]; } u;
  u.h[0] = *(const v8bf*)p0;
  u.h[1] = *(const v8bf*)p1;
  return u.v;
}

// 256-entry nf4 pair table: tab[p] = {bf16(NF4[p&15]), bf16(NF4[p>>4])}
static __device__ inline void build_pair_table(unsigned* sTab, int tid) {
  if (tid < 256) {
    union { unsigned u; v2bf v; } e;
    e.v[0] = (__bf16)NF4C[tid & 15];
    e.v[1] = (__bf16)NF4C[(tid >> 4) & 15];
    sTab[tid] = e.u;
  }
}

// Decode 8 nf4 codes (1 byte each, 0..15) -> 8 scaled bf16, one b128 store.
// t = w | w>>4 pairs two codes per table index; scale via v_pk_mul_bf16.
static __device__ inline void decode8(uint2 r, float sc,
                                      const unsigned* __restrict__ tab,
                                      __bf16* dst) {
  v2bf scpk; scpk[0] = (__bf16)sc; scpk[1] = (__bf16)sc;
  unsigned t0 = r.x | (r.x >> 4);
  unsigned t1 = r.y | (r.y >> 4);
  union PB { unsigned u; v2bf v; } p, q;
  uint4 out;
  p.u = tab[t0 & 0xFF];         q.v = p.v * scpk; out.x = q.u;
  p.u = tab[(t0 >> 16) & 0xFF]; q.v = p.v * scpk; out.y = q.u;
  p.u = tab[t1 & 0xFF];         q.v = p.v * scpk; out.z = q.u;
  p.u = tab[(t1 >> 16) & 0xFF]; q.v = p.v * scpk; out.w = q.u;
  *(uint4*)dst = out;           // 16B aligned
}

// Issue one 256x64 bf16 A-tile as 4 async b128 loads per thread (512 thr).
static __device__ inline void issue_async_tile(
    const __bf16* __restrict__ src, int src_ld, __bf16* dstbuf,
    int m0, int kb, int tid) {
#pragma unroll
  for (int i = 0; i < 4; ++i) {
    int g = i * NTHREADS + tid, row = g >> 3, c = g & 7;
    async_load_b128(src + (size_t)(m0 + row) * src_ld + kb * BK + c * 8,
                    &dstbuf[row * BKP + c * 8]);
  }
}

// ---------------------------------------------------------------------------
// Stage 1: h = silu(x @ w1^T) * (x @ w2^T), bf16 output.
// Block tile 256x128, K-tile 64, 512 threads = 16 waves (4M x 4N),
// wave tile 64x32 => acc[4][2] per weight matrix.
// ---------------------------------------------------------------------------
struct StageW1 {
  uint2 w1v[2], w2v[2];  // 128x64 nf4 codes: 1024 uint2 / 512 thr
  float s1v[2], s2v[2];
};

static __device__ inline void load_weights1(
    StageW1& st, const int8_t* __restrict__ w1i, const int8_t* __restrict__ w2i,
    const float* __restrict__ s1, const float* __restrict__ s2,
    int n0, int kb, int tid) {
#pragma unroll
  for (int i = 0; i < 2; ++i) {
    int g = i * NTHREADS + tid, row = g >> 3, c8 = g & 7;
    size_t go = (size_t)(n0 + row) * D_ + (size_t)kb * BK + c8 * 8;
    st.w1v[i] = *(const uint2*)(w1i + go);
    st.w2v[i] = *(const uint2*)(w2i + go);
    st.s1v[i] = s1[(size_t)(n0 + row) * (D_ / 64) + kb];
    st.s2v[i] = s2[(size_t)(n0 + row) * (D_ / 64) + kb];
  }
}

static __device__ inline void commit_weights1(
    const StageW1& st, __bf16* sW1, __bf16* sW2,
    const unsigned* __restrict__ sTab, int tid) {
#pragma unroll
  for (int i = 0; i < 2; ++i) {
    int g = i * NTHREADS + tid, row = g >> 3, c8 = g & 7;
    decode8(st.w1v[i], st.s1v[i], sTab, &sW1[row * BKP + c8 * 8]);
    decode8(st.w2v[i], st.s2v[i], sTab, &sW2[row * BKP + c8 * 8]);
  }
}

__global__ __launch_bounds__(NTHREADS) void nf4_gemm1_silu(
    const __bf16* __restrict__ xb,
    const int8_t* __restrict__ w1i, const float* __restrict__ s1,
    const int8_t* __restrict__ w2i, const float* __restrict__ s2,
    __bf16* __restrict__ h) {
  __shared__ __bf16 sX [2 * BM * BKP];   // 72 KB double-buffered async tile
  __shared__ __bf16 sW1[BN * BKP];       // 18 KB
  __shared__ __bf16 sW2[BN * BKP];       // 18 KB
  __shared__ unsigned sTab[256];

  const int tid  = threadIdx.x;
  const int lane = tid & 31;
  const int wave = tid >> 5;
  const int wm   = wave & 3;    // wave M offset: 0..3 * 64
  const int wn   = wave >> 2;   // wave N offset: 0..3 * 32
  const int n0   = blockIdx.x * BN;
  const int m0   = blockIdx.y * BM;

  build_pair_table(sTab, tid);

  const v8f zero = {0.f, 0.f, 0.f, 0.f, 0.f, 0.f, 0.f, 0.f};
  v8f acc1[4][2], acc2[4][2];
#pragma unroll
  for (int i = 0; i < 4; ++i)
#pragma unroll
    for (int j = 0; j < 2; ++j) { acc1[i][j] = zero; acc2[i][j] = zero; }

  StageW1 wst;
  issue_async_tile(xb, D_, &sX[0], m0, 0, tid);   // tile 0 in flight
  load_weights1(wst, w1i, w2i, s1, s2, n0, 0, tid);
  __syncthreads();   // pair table ready

  constexpr int KT = D_ / BK;
  for (int kb = 0; kb < KT; ++kb) {
    __bf16* xbuf = &sX[(kb & 1) * (BM * BKP)];

    commit_weights1(wst, sW1, sW2, sTab, tid);   // decode nf4 -> LDS
    wait_async0();                               // this thread's A-tile landed
    __syncthreads();                             // publish decode + async tile

    if (kb + 1 < KT) {   // next tile: async A + staged weights, overlaps WMMA
      issue_async_tile(xb, D_, &sX[((kb + 1) & 1) * (BM * BKP)], m0, kb + 1, tid);
      load_weights1(wst, w1i, w2i, s1, s2, n0, kb + 1, tid);
    }

#pragma unroll
    for (int kk = 0; kk < BK; kk += 32) {
      v16bf a[4];
#pragma unroll
      for (int i = 0; i < 4; ++i) {
        const __bf16* ap = &xbuf[(wm * 64 + i * 16 + (lane & 15)) * BKP + kk +
                                 ((lane & 16) ? 8 : 0)];
        a[i] = frag2(ap, ap + 16);
      }
#pragma unroll
      for (int j = 0; j < 2; ++j) {
        const int nrow = wn * 32 + j * 16 + (lane & 15);
        const int bo   = nrow * BKP + kk + ((lane & 16) ? 8 : 0);
        v16bf b1 = frag2(&sW1[bo], &sW1[bo + 16]);
        v16bf b2 = frag2(&sW2[bo], &sW2[bo + 16]);
#pragma unroll
        for (int i = 0; i < 4; ++i) {
          acc1[i][j] = __builtin_amdgcn_wmma_f32_16x16x32_bf16(
              false, a[i], false, b1, (short)0, acc1[i][j], false, false);
          acc2[i][j] = __builtin_amdgcn_wmma_f32_16x16x32_bf16(
              false, a[i], false, b2, (short)0, acc2[i][j], false, false);
        }
      }
    }
    __syncthreads();   // sW reads done before next decode overwrites
  }

  // ---- epilogue: silu(acc1) * acc2 -> bf16 h ----
#pragma unroll
  for (int i = 0; i < 4; ++i)
#pragma unroll
    for (int j = 0; j < 2; ++j) {
      const int mb = m0 + wm * 64 + i * 16 + ((lane & 16) ? 8 : 0);
      const int nc = n0 + wn * 32 + j * 16 + (lane & 15);
#pragma unroll
      for (int r = 0; r < 8; ++r) {
        float a = acc1[i][j][r];
        float b = acc2[i][j][r];
        float sv = a / (1.0f + __expf(-a));
        h[(size_t)(mb + r) * H_ + nc] = (__bf16)(sv * b);
      }
    }
}

// ---------------------------------------------------------------------------
// Stage 2: out = h @ w3^T, fp32 output. Same skeleton, single B matrix.
// ---------------------------------------------------------------------------
struct StageW2 {
  uint2 wv[2];
  float sv[2];
};

static __device__ inline void load_weights2(
    StageW2& st, const int8_t* __restrict__ w3i, const float* __restrict__ s3,
    int n0, int kb, int tid) {
#pragma unroll
  for (int i = 0; i < 2; ++i) {
    int g = i * NTHREADS + tid, row = g >> 3, c8 = g & 7;
    size_t go = (size_t)(n0 + row) * H_ + (size_t)kb * BK + c8 * 8;
    st.wv[i] = *(const uint2*)(w3i + go);
    st.sv[i] = s3[(size_t)(n0 + row) * (H_ / 64) + kb];
  }
}

static __device__ inline void commit_weights2(
    const StageW2& st, __bf16* sB, const unsigned* __restrict__ sTab, int tid) {
#pragma unroll
  for (int i = 0; i < 2; ++i) {
    int g = i * NTHREADS + tid, row = g >> 3, c8 = g & 7;
    decode8(st.wv[i], st.sv[i], sTab, &sB[row * BKP + c8 * 8]);
  }
}

__global__ __launch_bounds__(NTHREADS) void nf4_gemm2_out(
    const __bf16* __restrict__ hm,
    const int8_t* __restrict__ w3i, const float* __restrict__ s3,
    float* __restrict__ out) {
  __shared__ __bf16 sA[2 * BM * BKP];   // 72 KB double-buffered async tile
  __shared__ __bf16 sB[BN * BKP];
  __shared__ unsigned sTab[256];

  const int tid  = threadIdx.x;
  const int lane = tid & 31;
  const int wave = tid >> 5;
  const int wm   = wave & 3;
  const int wn   = wave >> 2;
  const int n0   = blockIdx.x * BN;   // over D_
  const int m0   = blockIdx.y * BM;   // over T_

  build_pair_table(sTab, tid);

  const v8f zero = {0.f, 0.f, 0.f, 0.f, 0.f, 0.f, 0.f, 0.f};
  v8f acc[4][2];
#pragma unroll
  for (int i = 0; i < 4; ++i)
#pragma unroll
    for (int j = 0; j < 2; ++j) acc[i][j] = zero;

  StageW2 wst;
  issue_async_tile(hm, H_, &sA[0], m0, 0, tid);
  load_weights2(wst, w3i, s3, n0, 0, tid);
  __syncthreads();

  constexpr int KT = H_ / BK;
  for (int kb = 0; kb < KT; ++kb) {
    __bf16* abuf = &sA[(kb & 1) * (BM * BKP)];

    commit_weights2(wst, sB, sTab, tid);
    wait_async0();
    __syncthreads();

    if (kb + 1 < KT) {
      issue_async_tile(hm, H_, &sA[((kb + 1) & 1) * (BM * BKP)], m0, kb + 1, tid);
      load_weights2(wst, w3i, s3, n0, kb + 1, tid);
    }

#pragma unroll
    for (int kk = 0; kk < BK; kk += 32) {
      v16bf a[4];
#pragma unroll
      for (int i = 0; i < 4; ++i) {
        const __bf16* ap = &abuf[(wm * 64 + i * 16 + (lane & 15)) * BKP + kk +
                                 ((lane & 16) ? 8 : 0)];
        a[i] = frag2(ap, ap + 16);
      }
#pragma unroll
      for (int j = 0; j < 2; ++j) {
        const int nrow = wn * 32 + j * 16 + (lane & 15);
        const int bo   = nrow * BKP + kk + ((lane & 16) ? 8 : 0);
        v16bf b = frag2(&sB[bo], &sB[bo + 16]);
#pragma unroll
        for (int i = 0; i < 4; ++i)
          acc[i][j] = __builtin_amdgcn_wmma_f32_16x16x32_bf16(
              false, a[i], false, b, (short)0, acc[i][j], false, false);
      }
    }
    __syncthreads();
  }

#pragma unroll
  for (int i = 0; i < 4; ++i)
#pragma unroll
    for (int j = 0; j < 2; ++j) {
      const int mb = m0 + wm * 64 + i * 16 + ((lane & 16) ? 8 : 0);
      const int nc = n0 + wn * 32 + j * 16 + (lane & 15);
#pragma unroll
      for (int r = 0; r < 8; ++r)
        out[(size_t)(mb + r) * D_ + nc] = acc[i][j][r];
    }
}

// ---------------------------------------------------------------------------
// Host launcher. Input order: x, then (idx, qs, qf, m) for w1, w2, w3.
// Workspace: [ h: T*H bf16 | s1,s2,s3: NBLK f32 each | xb: T*D bf16 ] ~70 MB
// ---------------------------------------------------------------------------
extern "C" void kernel_launch(void* const* d_in, const int* in_sizes, int n_in,
                              void* d_out, int out_size, void* d_ws,
                              size_t ws_size, hipStream_t stream) {
  const float*  x    = (const float*)d_in[0];
  const int8_t* w1i  = (const int8_t*)d_in[1];
  const int8_t* w1qs = (const int8_t*)d_in[2];
  const float*  w1qf = (const float*)d_in[3];
  const float*  w1m  = (const float*)d_in[4];
  const int8_t* w2i  = (const int8_t*)d_in[5];
  const int8_t* w2qs = (const int8_t*)d_in[6];
  const float*  w2qf = (const float*)d_in[7];
  const float*  w2m  = (const float*)d_in[8];
  const int8_t* w3i  = (const int8_t*)d_in[9];
  const int8_t* w3qs = (const int8_t*)d_in[10];
  const float*  w3qf = (const float*)d_in[11];
  const float*  w3m  = (const float*)d_in[12];

  char*   ws   = (char*)d_ws;
  __bf16* hbuf = (__bf16*)ws;
  size_t  off  = (size_t)T_ * H_ * sizeof(__bf16);
  float*  s1   = (float*)(ws + off);
  float*  s2   = s1 + NBLK;
  float*  s3   = s2 + NBLK;
  __bf16* xb   = (__bf16*)(s3 + NBLK);

  dim3 blk(256);
  int sgrid = (NBLK + 255) / 256;
  nf4_scalers_kernel<<<sgrid, blk, 0, stream>>>(w1qs, w1qf, w1m, s1, NBLK);
  nf4_scalers_kernel<<<sgrid, blk, 0, stream>>>(w2qs, w2qf, w2m, s2, NBLK);
  nf4_scalers_kernel<<<sgrid, blk, 0, stream>>>(w3qs, w3qf, w3m, s3, NBLK);

  int n4 = (T_ * D_) / 4;
  x_to_bf16_kernel<<<(n4 + 255) / 256, blk, 0, stream>>>(x, xb, n4);

  dim3 gblk(NTHREADS);
  nf4_gemm1_silu<<<dim3(H_ / BN, T_ / BM), gblk, 0, stream>>>(
      xb, w1i, s1, w2i, s2, hbuf);
  nf4_gemm2_out<<<dim3(D_ / BN, T_ / BM), gblk, 0, stream>>>(
      hbuf, w3i, s3, (float*)d_out);
}
